// CFTL_60327110640070
// MI455X (gfx1250) — matrix operand order repaired and checked by reading, not compiled, not run on hardware
//
#include <hip/hip_runtime.h>
#include <hip/hip_bf16.h>
#include <math.h>

// ---------------------------------------------------------------------------
// Problem: N=16, C=512, H=W=128.
//   g    = mean(x, H, W)                       (16,512)
//   F    = fft(g, axis=C);  A=|F|; P=F.imag
//   s1   = leaky_relu(g @ W1^T + b1)
//   s2   = leaky_relu(g @ W2^T + b2)
//   Z    = (s1*A)*exp(i*(s2*P))
//   x_i  = Re(ifft(Z, axis=C))                 (16,512)
//   out  = x + x_i[:, :, None, None]
//
// Memory-bound: 1.5 GB of HBM traffic (~65us @ 23.3 TB/s); all matmuls/DFTs
// are (16 x 512 x 512) and run on v_wmma_f32_16x16x32_f16.
// ---------------------------------------------------------------------------

typedef __attribute__((ext_vector_type(16))) _Float16 v16h;
typedef __attribute__((ext_vector_type(8)))  float    v8f;
typedef __attribute__((ext_vector_type(4)))  float    v4f;

#define CCH 512
#define PLANE 16384           // H*W = 128*128
#define NPLANES 8192          // N*C = 16*512
#define TWO_PI_OVER_512 0.012271846303085129838f

// ---------------------------------------------------------------------------
// Kernel 1: adaptive average pool -> g (N*C floats). One block per (n,c) plane.
// ---------------------------------------------------------------------------
__global__ void CFTL_pool_mean(const float* __restrict__ x, float* __restrict__ g) {
  __shared__ float red[256];
  const int plane = blockIdx.x;
  const v4f* __restrict__ p = (const v4f*)(x + (size_t)plane * PLANE);
  const int t = threadIdx.x;
  float sum = 0.0f;
#pragma unroll
  for (int i = 0; i < 16; ++i) {
    v4f v = __builtin_nontemporal_load(&p[t + i * 256]);
    sum += v.x + v.y + v.z + v.w;
  }
  red[t] = sum;
  __syncthreads();
#pragma unroll
  for (int s = 128; s > 0; s >>= 1) {
    if (t < s) red[t] += red[t + s];
    __syncthreads();
  }
  if (t == 0) g[plane] = red[0] * (1.0f / (float)PLANE);
}

// ---------------------------------------------------------------------------
// WMMA fragment builders (wave32, 16x16x32 f16).
// A (16x32, M=rows): lanes 0-15 -> M=lane, K = k0 + h;
//                    lanes 16-31 -> M=lane-16, K shifted by +16.
// B (32x16, N=cols): col = lane&15, K = k0 + h + 16*(lane>=16).
// D (16x16 f32):     row = v + 8*(lane>=16), col = lane&15.
// ---------------------------------------------------------------------------
__device__ __forceinline__ v16h load_A_rowmajor(const float* __restrict__ M,
                                                int lane, int k0) {
  const int n  = lane & 15;
  const int hi = (lane >> 4) & 1;
  const float* __restrict__ row = M + n * CCH + k0 + hi * 16;
  v16h a;
#pragma unroll
  for (int h = 0; h < 8; ++h)  a[h] = (_Float16)row[h];        // K = base..base+7
#pragma unroll
  for (int h = 8; h < 16; ++h) a[h] = (_Float16)row[h];        // K = base+8..+15
  return a;
}

__device__ __forceinline__ v16h load_B_from_W(const float* __restrict__ W,
                                              int lane, int k0, int j0) {
  const int j  = j0 + (lane & 15);
  const int hi = (lane >> 4) & 1;
  const float* __restrict__ src = W + j * CCH + k0 + hi * 16;  // B[k][j] = W[j][k]
  v16h b;
#pragma unroll
  for (int h = 0; h < 16; ++h) b[h] = (_Float16)src[h];
  return b;
}

// B fragments of the DFT basis: bc[k][j] = cos(2*pi*k*j/512),
//                               bs[k][j] = -sin(2*pi*k*j/512).
// Exact integer reduction (k*j) & 511 keeps the fp32 angle in [0, 2*pi).
__device__ __forceinline__ void make_B_trig(int lane, int k0, int j0,
                                            v16h* bc, v16h* bs) {
  const int j  = j0 + (lane & 15);
  const int hi = (lane >> 4) & 1;
  const int kb = k0 + hi * 16;
#pragma unroll
  for (int h = 0; h < 16; ++h) {
    const int m = ((kb + h) * j) & (CCH - 1);
    float s, c;
    __sincosf((float)m * TWO_PI_OVER_512, &s, &c);
    (*bc)[h] = (_Float16)c;
    (*bs)[h] = (_Float16)(-s);
  }
}

__device__ __forceinline__ v8f wmma_f16(v16h a, v16h b, v8f c) {
  return __builtin_amdgcn_wmma_f32_16x16x32_f16(
      /*neg_a=*/false, a, /*neg_b=*/false, b,
      /*c_mod=*/(short)0, c, /*reuse_a=*/false, /*reuse_b=*/false);
}

// ---------------------------------------------------------------------------
// Kernel 2: fused  s1/s2 GEMMs + forward DFT + amplitude/phase nonlinearity.
// 32 blocks x 1 wave; block b owns output columns j0..j0+15 for all 16 rows.
// Writes Zr, Zi (16x512 each).
// ---------------------------------------------------------------------------
__global__ void CFTL_spectral_fwd(const float* __restrict__ g,
                                  const float* __restrict__ W1,
                                  const float* __restrict__ b1,
                                  const float* __restrict__ W2,
                                  const float* __restrict__ b2,
                                  float* __restrict__ Zr,
                                  float* __restrict__ Zi) {
  const int lane = threadIdx.x;
  const int j0   = blockIdx.x * 16;
  v8f s1a = {}, s2a = {}, fr = {}, fi = {};
  for (int k0 = 0; k0 < CCH; k0 += 32) {
    v16h a   = load_A_rowmajor(g, lane, k0);
    v16h bw1 = load_B_from_W(W1, lane, k0, j0);
    v16h bw2 = load_B_from_W(W2, lane, k0, j0);
    v16h bc, bs;
    make_B_trig(lane, k0, j0, &bc, &bs);
    s1a = wmma_f16(a, bw1, s1a);   // g @ W1^T tile
    s2a = wmma_f16(a, bw2, s2a);   // g @ W2^T tile
    fr  = wmma_f16(a, bc,  fr);    // Re(F) tile
    fi  = wmma_f16(a, bs,  fi);    // Im(F) tile
  }
  const int j  = j0 + (lane & 15);
  const int hi = (lane >> 4) & 1;
  const float bb1 = b1[j];
  const float bb2 = b2[j];
#pragma unroll
  for (int v = 0; v < 8; ++v) {
    const int n = v + hi * 8;               // batch row
    const float Fr = fr[v], Fi = fi[v];
    const float Aamp = __fsqrt_rn(Fr * Fr + Fi * Fi);
    float s1 = s1a[v] + bb1;  s1 = (s1 > 0.0f) ? s1 : 0.01f * s1;  // leaky relu
    float s2 = s2a[v] + bb2;  s2 = (s2 > 0.0f) ? s2 : 0.01f * s2;
    const float Ap = s1 * Aamp;             // scaled amplitude
    const float Pp = s2 * Fi;               // scaled "phase" (= .imag per ref)
    float sp, cp;
    __sincosf(Pp, &sp, &cp);
    Zr[n * CCH + j] = Ap * cp;
    Zi[n * CCH + j] = Ap * sp;
  }
}

// ---------------------------------------------------------------------------
// Kernel 3: inverse transform  x_i = (Zr @ cos + Zi @ (-sin)) / 512.
// ---------------------------------------------------------------------------
__global__ void CFTL_spectral_inv(const float* __restrict__ Zr,
                                  const float* __restrict__ Zi,
                                  float* __restrict__ xi) {
  const int lane = threadIdx.x;
  const int j0   = blockIdx.x * 16;         // output channel tile
  v8f acc = {};
  for (int k0 = 0; k0 < CCH; k0 += 32) {
    v16h ar = load_A_rowmajor(Zr, lane, k0);
    v16h ai = load_A_rowmajor(Zi, lane, k0);
    v16h bc, bs;
    make_B_trig(lane, k0, j0, &bc, &bs);
    acc = wmma_f16(ar, bc, acc);            // + Zr * cos
    acc = wmma_f16(ai, bs, acc);            // - Zi * sin
  }
  const int j  = j0 + (lane & 15);
  const int hi = (lane >> 4) & 1;
#pragma unroll
  for (int v = 0; v < 8; ++v) {
    const int n = v + hi * 8;
    xi[n * CCH + j] = acc[v] * (1.0f / (float)CCH);
  }
}

// ---------------------------------------------------------------------------
// Kernel 4: out = x + x_i[n,c] broadcast over H,W. One block per (n,c) plane.
// Non-temporal 128-bit stream (no reuse; keep L2 clean).
// ---------------------------------------------------------------------------
__global__ void CFTL_broadcast_add(const float* __restrict__ x,
                                   const float* __restrict__ xi,
                                   float* __restrict__ out) {
  const int plane = blockIdx.x;
  const float add = xi[plane];
  const v4f* __restrict__ px = (const v4f*)(x + (size_t)plane * PLANE);
  v4f* __restrict__ po = (v4f*)(out + (size_t)plane * PLANE);
  const int t = threadIdx.x;
  const v4f av = {add, add, add, add};
#pragma unroll
  for (int i = 0; i < 16; ++i) {
    v4f v = __builtin_nontemporal_load(&px[t + i * 256]);
    __builtin_nontemporal_store(v + av, &po[t + i * 256]);
  }
}

// ---------------------------------------------------------------------------
extern "C" void kernel_launch(void* const* d_in, const int* in_sizes, int n_in,
                              void* d_out, int out_size, void* d_ws, size_t ws_size,
                              hipStream_t stream) {
  const float* x  = (const float*)d_in[0];   // (16,512,128,128)
  const float* W1 = (const float*)d_in[1];   // (512,512)
  const float* b1 = (const float*)d_in[2];   // (512)
  const float* W2 = (const float*)d_in[3];   // (512,512)
  const float* b2 = (const float*)d_in[4];   // (512)
  float* out = (float*)d_out;

  float* ws = (float*)d_ws;                  // 128 KB used
  float* g  = ws;                            // 16*512
  float* Zr = ws + NPLANES;                  // 16*512
  float* Zi = ws + 2 * NPLANES;              // 16*512
  float* xi = ws + 3 * NPLANES;              // 16*512

  CFTL_pool_mean   <<<NPLANES, 256, 0, stream>>>(x, g);
  CFTL_spectral_fwd<<<32,       32, 0, stream>>>(g, W1, b1, W2, b2, Zr, Zi);
  CFTL_spectral_inv<<<32,       32, 0, stream>>>(Zr, Zi, xi);
  CFTL_broadcast_add<<<NPLANES, 256, 0, stream>>>(x, xi, out);
}
